// SparseAutoencoder_12249246728715
// MI455X (gfx1250) — matrix operand order repaired and checked by reading, not compiled, not run on hardware
//
#include <hip/hip_runtime.h>
#include <hip/hip_bf16.h>

typedef __attribute__((ext_vector_type(16))) _Float16 v16h;
typedef __attribute__((ext_vector_type(8)))  _Float16 v8h;
typedef __attribute__((ext_vector_type(8)))  float    v8f;
typedef __attribute__((ext_vector_type(4)))  int      v4i;

#define N_ROWS   32768   // 16*2048
#define C_DIM    1024
#define H_DIM    2048
#define TOPK     256

#define B_ROW_BYTES 80                    // 64B payload + 16B pad (bank-conflict-free)
#define B_BUF_BYTES (64 * B_ROW_BYTES)    // one 64x32 f16 tile, padded

// ---- gfx1250 async global->LDS path (guarded; sync fallback keeps build green) ----
#if defined(__has_builtin)
#if __has_builtin(__builtin_amdgcn_global_load_async_to_lds_b128) && \
    __has_builtin(__builtin_amdgcn_s_wait_asynccnt)
#define HAVE_ASYNC_LDS 1
#endif
#if __has_builtin(__builtin_amdgcn_s_barrier_signal) && \
    __has_builtin(__builtin_amdgcn_s_barrier_wait) && \
    __has_builtin(__builtin_amdgcn_s_wait_dscnt)
#define HAVE_SPLIT_BARRIER 1
#endif
#endif

#define AS1 __attribute__((address_space(1)))
#define AS3 __attribute__((address_space(3)))

typedef AS1 v4i* gv4i_ptr;
typedef AS3 v4i* lv4i_ptr;

__device__ __forceinline__ void copy_b128_to_lds(const void* g, void* l) {
#ifdef HAVE_ASYNC_LDS
    __builtin_amdgcn_global_load_async_to_lds_b128(
        (gv4i_ptr)g, (lv4i_ptr)l, /*offset=*/0, /*cpol=*/0);
#else
    *(uint4*)l = *(const uint4*)g;
#endif
}

template <int N>
__device__ __forceinline__ void wait_async_le() {
#ifdef HAVE_ASYNC_LDS
    __builtin_amdgcn_s_wait_asynccnt((short)N);
#endif
}

// LDS-only barrier: drain DScnt, then split workgroup barrier.
// Does NOT drain global LOADcnt, so A-operand prefetch loads stay in flight.
__device__ __forceinline__ void lds_barrier() {
#ifdef HAVE_SPLIT_BARRIER
    __builtin_amdgcn_s_wait_dscnt(0);
    __builtin_amdgcn_s_barrier_signal(-1);
    __builtin_amdgcn_s_barrier_wait(-1);
#else
    __syncthreads();
#endif
}

// ---------------------------------------------------------------------------
// Kernel 1: W (f32, [H][C]) -> W16 ([H][C] f16) and Wt16 ([C][H] f16).
// Also zero the loss accumulators (harness does not re-poison between replays).
// ---------------------------------------------------------------------------
__global__ __launch_bounds__(256) void sae_convert_w(const float* __restrict__ W,
                                                     _Float16* __restrict__ W16,
                                                     _Float16* __restrict__ Wt16,
                                                     float* __restrict__ accum) {
    int i = blockIdx.x * 256 + threadIdx.x;          // 0 .. H*C-1
    if (i == 0) { accum[0] = 0.0f; accum[1] = 0.0f; }
    int h = i >> 10;           // /C_DIM
    int c = i & (C_DIM - 1);
    _Float16 v = (_Float16)W[i];
    W16[i] = v;
    Wt16[(size_t)c * H_DIM + h] = v;
}

// ---- A operand: f32 row -> f16 16x32 WMMA A layout (per-lane two 8-K chunks) ----
struct A4 { float4 q0, q1, q2, q3; };

__device__ __forceinline__ A4 load_a(const float* __restrict__ rowp, int abase) {
    A4 r;
    r.q0 = *(const float4*)(rowp + abase);
    r.q1 = *(const float4*)(rowp + abase + 4);
    r.q2 = *(const float4*)(rowp + abase + 16);
    r.q3 = *(const float4*)(rowp + abase + 20);
    return r;
}

__device__ __forceinline__ v16h cvt_a(A4 q) {
    v16h a;
    a[0]  = (_Float16)q.q0.x; a[1]  = (_Float16)q.q0.y; a[2]  = (_Float16)q.q0.z; a[3]  = (_Float16)q.q0.w;
    a[4]  = (_Float16)q.q1.x; a[5]  = (_Float16)q.q1.y; a[6]  = (_Float16)q.q1.z; a[7]  = (_Float16)q.q1.w;
    a[8]  = (_Float16)q.q2.x; a[9]  = (_Float16)q.q2.y; a[10] = (_Float16)q.q2.z; a[11] = (_Float16)q.q2.w;
    a[12] = (_Float16)q.q3.x; a[13] = (_Float16)q.q3.y; a[14] = (_Float16)q.q3.z; a[15] = (_Float16)q.q3.w;
    return a;
}

// B operand from padded LDS tile: two 16B chunks 32 bytes apart at 16B alignment
__device__ __forceinline__ v16h ld_b_lds(const char* p) {
    v8h b0 = *(const v8h*)(p);
    v8h b1 = *(const v8h*)(p + 32);
    v16h b;
    #pragma unroll
    for (int i = 0; i < 8; ++i) { b[i] = b0[i]; b[8 + i] = b1[i]; }
    return b;
}

// Load all 4 B tiles first (distinct regs), then 4 back-to-back WMMAs:
// DS latency of tiles 1-3 hides under WMMAs 0-2.
__device__ __forceinline__ void wmma4(v8f acc[4], v16h a, const char* bufp, int boff) {
    v16h b0 = ld_b_lds(bufp + boff);
    v16h b1 = ld_b_lds(bufp + boff + 16 * B_ROW_BYTES);
    v16h b2 = ld_b_lds(bufp + boff + 32 * B_ROW_BYTES);
    v16h b3 = ld_b_lds(bufp + boff + 48 * B_ROW_BYTES);
    acc[0] = __builtin_amdgcn_wmma_f32_16x16x32_f16(false, a, false, b0, (short)0, acc[0], false, false);
    acc[1] = __builtin_amdgcn_wmma_f32_16x16x32_f16(false, a, false, b1, (short)0, acc[1], false, false);
    acc[2] = __builtin_amdgcn_wmma_f32_16x16x32_f16(false, a, false, b2, (short)0, acc[2], false, false);
    acc[3] = __builtin_amdgcn_wmma_f32_16x16x32_f16(false, a, false, b3, (short)0, acc[3], false, false);
}

// ---------------------------------------------------------------------------
// Generic 128(M) x 64(N) block GEMM: out = cvt16(Af32) @ B16^layout.
//   A: f32 [*, Klen] rows (x or latent), converted on the fly.
//   B: f16 [Ncols][Klen] row-major (W16 or Wt16) -> columns of B operand.
// 8 waves/block, each wave a 16x64 tile. B tile (64x32 f16 = 4KB) staged in
// TRIPLE-buffered LDS via async global->LDS copies shared by the whole block:
// one barrier per K-step. Copy for step i+2 is issued only after the ready-
// barrier of step i (which proves all waves finished reading buffer (i-1)%3,
// exactly the buffer being overwritten). ASYNCcnt: <=2 in flight, wait <=1
// per step; last two steps peeled (wait <=1, then wait <=0).
// ---------------------------------------------------------------------------
template <int KLEN, int ASTRIDE>
__device__ __forceinline__ void block_gemm_f16(const float* __restrict__ Arows,
                                               const _Float16* __restrict__ Bmat,
                                               int m0_wave_row, int n0, int tid,
                                               v8f acc[4], char* smemB) {
    const int lane  = tid & 31;
    const int lr    = lane & 15;
    const int ahalf = (lane >> 4) * 8;          // A K-offset for upper half-wave
    const int bbyte = (lane & 16) ? 16 : 0;     // B K-offset (bytes) for upper half-wave
    const int boff  = lr * B_ROW_BYTES + bbyte;

    const float* xrow = Arows + (size_t)(m0_wave_row + lr) * ASTRIDE;

    // cooperative B copy: thread t moves 16B: row = t/4 (N), chunk = t%4 (K)
    const int crow = tid >> 2, cchk = tid & 3;
    const _Float16* gB = Bmat + (size_t)(n0 + crow) * KLEN + cchk * 8;
    char* lB = smemB + crow * B_ROW_BYTES + cchk * 16;

    constexpr int NSTEP = KLEN / 32;            // >= 3
    static_assert(NSTEP >= 3, "need at least 3 K-steps");

    copy_b128_to_lds(gB,      lB + 0 * B_BUF_BYTES);   // B(0) -> buf0
    copy_b128_to_lds(gB + 32, lB + 1 * B_BUF_BYTES);   // B(1) -> buf1
    A4 qa = load_a(xrow, ahalf);                       // A(0)

    int cur = 0;                                // buffer holding B(i)
    int tgt = 2;                                // buffer for B(i+2)
    #pragma unroll 3
    for (int i = 0; i < NSTEP - 2; ++i) {
        wait_async_le<1>();                     // B(i) landed (B(i+1) may be in flight)
        lds_barrier();                          // ready + (i-1)%3 fully consumed
        copy_b128_to_lds(gB + (i + 2) * 32, lB + tgt * B_BUF_BYTES);
        A4 qn = load_a(xrow, (i + 1) * 32 + ahalf);    // prefetch A(i+1)
        wmma4(acc, cvt_a(qa), smemB + cur * B_BUF_BYTES, boff);
        qa = qn;
        cur = (cur == 2) ? 0 : cur + 1;
        tgt = (tgt == 2) ? 0 : tgt + 1;
    }

    // step NSTEP-2: nothing new to issue
    {
        constexpr int c1 = (NSTEP - 2) % 3;
        wait_async_le<1>();                     // B(NSTEP-2) landed
        lds_barrier();
        A4 qn = load_a(xrow, (NSTEP - 1) * 32 + ahalf);
        wmma4(acc, cvt_a(qa), smemB + c1 * B_BUF_BYTES, boff);
        qa = qn;
    }
    // step NSTEP-1
    {
        constexpr int c2 = (NSTEP - 1) % 3;
        wait_async_le<0>();                     // B(NSTEP-1) landed
        lds_barrier();
        wmma4(acc, cvt_a(qa), smemB + c2 * B_BUF_BYTES, boff);
    }
}

// ---------------------------------------------------------------------------
// Kernel 2: enc = clip(x @ W^T + b_enc, -10, 10) -> latent buffer.
// Block: 128 rows x 64 H-cols. Consecutive blocks share the x row-stripe (L2).
// ---------------------------------------------------------------------------
__global__ __launch_bounds__(256) void sae_encode(const float* __restrict__ x,
                                                  const _Float16* __restrict__ W16,
                                                  const float* __restrict__ b_enc,
                                                  float* __restrict__ latent) {
    __shared__ __align__(32) char smemB[3 * B_BUF_BYTES];
    const int tid  = threadIdx.x;
    const int lane = tid & 31;
    const int wave = tid >> 5;
    const int m0 = (blockIdx.x >> 5) * 128 + wave * 16;   // 256 m-blocks
    const int h0 = (blockIdx.x & 31) * 64;                // 32 h-blocks

    v8f zero = {0.f,0.f,0.f,0.f,0.f,0.f,0.f,0.f};
    v8f acc[4] = {zero, zero, zero, zero};

    block_gemm_f16<C_DIM, C_DIM>(x, W16, m0, h0, tid, acc, smemB);

    const int lr   = lane & 15;
    const int mofs = (lane >> 4) * 8;
    #pragma unroll
    for (int j = 0; j < 4; ++j) {
        int n = h0 + 16 * j + lr;
        float be = b_enc[n];
        #pragma unroll
        for (int r = 0; r < 8; ++r) {
            int m = m0 + mofs + r;
            float v = acc[j][r] + be;
            v = fminf(fmaxf(v, -10.0f), 10.0f);
            latent[(size_t)m * H_DIM + n] = v;
        }
    }
}

// ---------------------------------------------------------------------------
// Kernel 3: per-row exact top-256 radix select (4x8-bit), zero rest, relu,
// rewrite in place, accumulate sum(|latent|). One 256-thread block per row.
// ---------------------------------------------------------------------------
__global__ __launch_bounds__(256) void sae_topk(float* __restrict__ latent,
                                                float* __restrict__ accum_sp) {
    __shared__ float    vals[H_DIM];
    __shared__ unsigned keys[H_DIM];
    __shared__ unsigned hist[256];
    __shared__ unsigned sscan[256];
    __shared__ unsigned s_prefix, s_rem;
    __shared__ float    s_red[8];

    const int row = blockIdx.x;
    const int t   = threadIdx.x;
    float* rowp = latent + (size_t)row * H_DIM;

    #pragma unroll
    for (int i = 0; i < 8; ++i) {
        int idx = t + 256 * i;
        float v = rowp[idx];
        vals[idx] = v;
        unsigned u = __float_as_uint(v);
        keys[idx] = (u & 0x80000000u) ? ~u : (u | 0x80000000u);  // order-preserving
    }
    if (t == 0) { s_prefix = 0u; s_rem = TOPK; }
    __syncthreads();

    for (int r = 0; r < 4; ++r) {
        hist[t] = 0u;
        __syncthreads();
        unsigned prefix = s_prefix;
        int hishift = 32 - 8 * r;
        int shift   = 24 - 8 * r;
        #pragma unroll
        for (int i = 0; i < 8; ++i) {
            unsigned key = keys[t + 256 * i];
            unsigned hb = (r == 0) ? 0u : (key >> hishift);
            if (hb == prefix) atomicAdd(&hist[(key >> shift) & 0xFF], 1u);
        }
        __syncthreads();
        sscan[t] = hist[t];                     // inclusive suffix-sum (Hillis-Steele)
        __syncthreads();
        for (int off = 1; off < 256; off <<= 1) {
            unsigned add = (t + off < 256) ? sscan[t + off] : 0u;
            __syncthreads();
            sscan[t] += add;
            __syncthreads();
        }
        unsigned rem = s_rem;
        unsigned sfx = sscan[t];
        unsigned nxt = (t < 255) ? sscan[t + 1] : 0u;
        if (sfx >= rem && nxt < rem) {          // unique crossing bin
            s_prefix = (prefix << 8) | (unsigned)t;
            s_rem    = rem - nxt;
        }
        __syncthreads();
    }

    const unsigned kth = s_prefix;              // key of the k-th largest element
    float lsum = 0.0f;
    #pragma unroll
    for (int i = 0; i < 8; ++i) {
        int idx = t + 256 * i;
        float out = (keys[idx] >= kth) ? fmaxf(vals[idx], 0.0f) : 0.0f;
        rowp[idx] = out;
        lsum += out;                            // |latent| == latent (>= 0)
    }
    #pragma unroll
    for (int off = 16; off > 0; off >>= 1) lsum += __shfl_down(lsum, off, 32);
    if ((t & 31) == 0) s_red[t >> 5] = lsum;
    __syncthreads();
    if (t == 0) {
        float v = 0.0f;
        #pragma unroll
        for (int i = 0; i < 8; ++i) v += s_red[i];
        atomicAdd(accum_sp, v);
    }
}

// ---------------------------------------------------------------------------
// Kernel 4: recon = latent @ W + b_dec ; fused sum((recon - x)^2).
// Block: 128 rows x 64 C-cols; B columns come from Wt16 [C][H].
// ---------------------------------------------------------------------------
__global__ __launch_bounds__(256) void sae_decode(const float* __restrict__ latent,
                                                  const _Float16* __restrict__ Wt16,
                                                  const float* __restrict__ b_dec,
                                                  const float* __restrict__ x,
                                                  float* __restrict__ recon,
                                                  float* __restrict__ accum_mse) {
    __shared__ __align__(32) char smemB[3 * B_BUF_BYTES];
    const int tid  = threadIdx.x;
    const int lane = tid & 31;
    const int wave = tid >> 5;
    const int m0 = (blockIdx.x >> 4) * 128 + wave * 16;   // 256 m-blocks
    const int c0 = (blockIdx.x & 15) * 64;                // 16 c-blocks

    v8f zero = {0.f,0.f,0.f,0.f,0.f,0.f,0.f,0.f};
    v8f acc[4] = {zero, zero, zero, zero};

    block_gemm_f16<H_DIM, H_DIM>(latent, Wt16, m0, c0, tid, acc, smemB);

    const int lr   = lane & 15;
    const int mofs = (lane >> 4) * 8;
    float msum = 0.0f;
    #pragma unroll
    for (int j = 0; j < 4; ++j) {
        int c = c0 + 16 * j + lr;
        float bd = b_dec[c];
        #pragma unroll
        for (int r = 0; r < 8; ++r) {
            int m = m0 + mofs + r;
            float v = acc[j][r] + bd;
            size_t off = (size_t)m * C_DIM + c;
            recon[off] = v;
            float d = v - x[off];
            msum += d * d;
        }
    }
    #pragma unroll
    for (int off = 16; off > 0; off >>= 1) msum += __shfl_down(msum, off, 32);
    if (lane == 0) atomicAdd(accum_mse, msum);
}

// ---------------------------------------------------------------------------
// Kernel 5: sae_loss = min(mse_mean,100) + 1e-3 * min(|latent|_mean,10)
// ---------------------------------------------------------------------------
__global__ void sae_finalize(const float* __restrict__ accum, float* __restrict__ loss) {
    float rl = accum[0] * (1.0f / ((float)N_ROWS * (float)C_DIM));
    rl = fminf(rl, 100.0f);
    float sp = accum[1] * (1.0f / ((float)N_ROWS * (float)H_DIM));
    sp = fminf(sp, 10.0f);
    *loss = rl + 0.001f * sp;
}

extern "C" void kernel_launch(void* const* d_in, const int* in_sizes, int n_in,
                              void* d_out, int out_size, void* d_ws, size_t ws_size,
                              hipStream_t stream) {
    const float* x     = (const float*)d_in[0];   // [16,2048,1024]
    const float* W     = (const float*)d_in[1];   // [2048,1024]
    const float* b_enc = (const float*)d_in[2];   // [2048]
    const float* b_dec = (const float*)d_in[3];   // [1024]

    float* recon  = (float*)d_out;                       // 33554432
    float* latent = recon + (size_t)N_ROWS * C_DIM;      // 67108864
    float* loss   = latent + (size_t)N_ROWS * H_DIM;     // 1

    char* ws = (char*)d_ws;
    float*    accum = (float*)ws;                                    // 2 floats
    _Float16* W16   = (_Float16*)(ws + 256);                         // 4 MB
    _Float16* Wt16  = (_Float16*)(ws + 256 + (size_t)H_DIM*C_DIM*2); // 4 MB

    sae_convert_w<<<(H_DIM * C_DIM) / 256, 256, 0, stream>>>(W, W16, Wt16, accum);
    sae_encode   <<<(N_ROWS / 128) * (H_DIM / 64), 256, 0, stream>>>(x, W16, b_enc, latent);
    sae_topk     <<<N_ROWS, 256, 0, stream>>>(latent, accum + 1);
    sae_decode   <<<(N_ROWS / 128) * (C_DIM / 64), 256, 0, stream>>>(latent, Wt16, b_dec, x,
                                                                     recon, accum);
    sae_finalize <<<1, 1, 0, stream>>>(accum, loss);
}
